// DeepLayer_68521908240969
// MI455X (gfx1250) — matrix operand order repaired and checked by reading, not compiled, not run on hardware
//
#include <hip/hip_runtime.h>
#include <math.h>

// ---------------------------------------------------------------------------
// Types for WMMA (CDNA5 gfx1250, wave32)
// ---------------------------------------------------------------------------
typedef __attribute__((ext_vector_type(16))) _Float16 v16h;
typedef __attribute__((ext_vector_type(8)))  float    v8f;

#define EPS_GEN 1e-7f
#define EPS_BN  1e-5f

// Async direct-to-LDS copy (CDNA5, ASYNCcnt-tracked).  lds = 32-bit LDS byte
// address (low 32 bits of generic shared pointer), ga = 64-bit global address.
#define ASYNC_B128(lds, ga)                                                  \
    asm volatile("global_load_async_to_lds_b128 %0, %1, off"                 \
                 :: "v"(lds), "v"(ga) : "memory")

// ---------------------------------------------------------------------------
// Utility kernels
// ---------------------------------------------------------------------------
__global__ void zero_kernel(float* __restrict__ p, size_t n) {
    size_t i = (size_t)blockIdx.x * blockDim.x + threadIdx.x;
    size_t stride = (size_t)gridDim.x * blockDim.x;
    for (; i < n; i += stride) p[i] = 0.0f;
}

// Wt[n*K + k] = (f16) W[k*N + n]   (weights are tiny: <= 256x256)
__global__ void transpose_f16_kernel(const float* __restrict__ W,
                                     _Float16* __restrict__ Wt, int K, int N) {
    int idx = blockIdx.x * blockDim.x + threadIdx.x;
    if (idx < K * N) {
        int n = idx / K;
        int k = idx - n * K;
        Wt[idx] = (_Float16)W[(size_t)k * N + n];
    }
}

// Column sums / sums-of-squares over an [M, W] matrix. blockDim.x == W.
__global__ void colstats_kernel(const float* __restrict__ X, int M, int W,
                                float* __restrict__ sums, float* __restrict__ sqs) {
    int t = threadIdx.x;
    float s = 0.0f, q = 0.0f;
    for (int r = blockIdx.x; r < M; r += gridDim.x) {
        float v = X[(size_t)r * W + t];
        s += v;
        q += v * v;
    }
    atomicAdd(&sums[t], s);
    atomicAdd(&sqs[t], q);
}

// scale = g * rsqrt(var + eps); shift = b - mean*scale.  blockDim.x == W.
__global__ void finalize_stats_kernel(const float* __restrict__ sums,
                                      const float* __restrict__ sqs,
                                      const float* __restrict__ g,
                                      const float* __restrict__ b,
                                      int M, float* __restrict__ scale,
                                      float* __restrict__ shift) {
    int t = threadIdx.x;
    float inv = 1.0f / (float)M;
    float mean = sums[t] * inv;
    float var  = sqs[t] * inv - mean * mean;
    float rs   = rsqrtf(var + EPS_BN);
    float sc   = g[t] * rs;
    scale[t] = sc;
    shift[t] = b[t] - mean * sc;
}

// f32 out: y = relu(x*scale[c]+shift[c])
__global__ void bn_relu_kernel(const float* __restrict__ X, float* __restrict__ Y,
                               const float* __restrict__ scale,
                               const float* __restrict__ shift,
                               size_t total, int wmask) {
    size_t i = (size_t)blockIdx.x * blockDim.x + threadIdx.x;
    size_t stride = (size_t)gridDim.x * blockDim.x;
    for (; i < total; i += stride) {
        int c = (int)(i & (size_t)wmask);
        float v = X[i] * scale[c] + shift[c];
        Y[i] = v > 0.0f ? v : 0.0f;
    }
}

// f16 out (feeds next WMMA GEMM)
__global__ void bn_relu_f16_kernel(const float* __restrict__ X,
                                   _Float16* __restrict__ Y,
                                   const float* __restrict__ scale,
                                   const float* __restrict__ shift,
                                   size_t total, int wmask) {
    size_t i = (size_t)blockIdx.x * blockDim.x + threadIdx.x;
    size_t stride = (size_t)gridDim.x * blockDim.x;
    for (; i < total; i += stride) {
        int c = (int)(i & (size_t)wmask);
        float v = X[i] * scale[c] + shift[c];
        Y[i] = (_Float16)(v > 0.0f ? v : 0.0f);
    }
}

// y16 = (f16)(t + x)   (residual, feeds encoder GEMM)
__global__ void add_f16_kernel(const float* __restrict__ T,
                               const float* __restrict__ X,
                               _Float16* __restrict__ Y, size_t n) {
    size_t i = (size_t)blockIdx.x * blockDim.x + threadIdx.x;
    size_t stride = (size_t)gridDim.x * blockDim.x;
    for (; i < n; i += stride) Y[i] = (_Float16)(T[i] + X[i]);
}

// ---------------------------------------------------------------------------
// Edge softmax aggregation.  msg = h[src] + eps (h >= 0 after relu).
// Pass A: segment max via int-bit atomicMax (all msgs > 0; init 0 matches the
//         reference's where(isfinite, mx, 0)).
// Pass B: den += exp(msg-mx[dst]); num += msg*exp(msg-mx[dst]).
// blockDim.x == C (128): coalesced 512B channel rows per edge.
// ---------------------------------------------------------------------------
__global__ void edge_max_kernel(const float* __restrict__ h,
                                const int* __restrict__ ei, int E,
                                int* __restrict__ mx) {
    int t = threadIdx.x;
    for (int e = blockIdx.x; e < E; e += gridDim.x) {
        int s = ei[e];
        int d = ei[E + e];
        float msg = h[(size_t)s * 128 + t] + EPS_GEN;
        atomicMax(&mx[(size_t)d * 128 + t], __float_as_int(msg));
    }
}

__global__ void edge_sum_kernel(const float* __restrict__ h,
                                const int* __restrict__ ei, int E,
                                const int* __restrict__ mx,
                                float* __restrict__ den,
                                float* __restrict__ num) {
    int t = threadIdx.x;
    for (int e = blockIdx.x; e < E; e += gridDim.x) {
        int s = ei[e];
        int d = ei[E + e];
        size_t di = (size_t)d * 128 + t;
        float msg = h[(size_t)s * 128 + t] + EPS_GEN;
        float m = __int_as_float(mx[di]);
        float ex = __expf(msg - m);
        atomicAdd(&den[di], ex);
        atomicAdd(&num[di], msg * ex);
    }
}

// out0 = num/(den+1e-16) + h, emitted as f16 (GEMM1 input)
__global__ void node_combine_f16_kernel(const float* __restrict__ h,
                                        const float* __restrict__ num,
                                        const float* __restrict__ den,
                                        _Float16* __restrict__ o16, size_t n) {
    size_t i = (size_t)blockIdx.x * blockDim.x + threadIdx.x;
    size_t stride = (size_t)gridDim.x * blockDim.x;
    for (; i < n; i += stride)
        o16[i] = (_Float16)(num[i] / (den[i] + 1e-16f) + h[i]);
}

// ---------------------------------------------------------------------------
// WMMA GEMM:  C[M,NCOLS] = A[M,K] @ Bt[NCOLS,K]^T + bias[NCOLS]
// A and Bt are f16 in memory; f32 accumulate; f32 output.
// Block: 256 threads = 8 wave32; tile 128(M) x 64(N); K-step 32.
// Double-buffered LDS filled with global_load_async_to_lds_b128 (ASYNCcnt):
// exactly 3 async b128 per thread per K-step (2 A-tile + 1 B-tile), pipelined
// one tile ahead with s_wait_asynccnt 3.  No conversion VALU, no LDS scatter.
// LDS rows padded to 40 halfs (80B): 16B alignment for b128, conflict-free
// fragment reads.  Requires K % 32 == 0, NCOLS % 64 == 0.
// ---------------------------------------------------------------------------
__global__ void __launch_bounds__(256)
gemm_wmma_kernel(const _Float16* __restrict__ A, const _Float16* __restrict__ Bt,
                 const float* __restrict__ bias, float* __restrict__ Cout,
                 int M, int K, int NCOLS) {
    __shared__ _Float16 AsAll[2 * 128 * 40];   // 2 x 10240 B
    __shared__ _Float16 BsAll[2 * 64 * 40];    // 2 x  5120 B

    const int tid = threadIdx.x;
    const int lane = tid & 31;
    const int w = tid >> 5;
    const int wm = w & 3;    // M sub-tile (32 rows)
    const int wn = w >> 2;   // N sub-tile (32 cols)
    const int l15 = lane & 15;
    const int hh = (lane >> 4) & 1;

    const int colBase = blockIdx.x * 64;
    const int rowBase = blockIdx.y * 128;

    v8f c00 = {}; v8f c01 = {}; v8f c10 = {}; v8f c11 = {};

    // Issue one tile's async copies (memory -> LDS, no VGPR staging).
    auto issue_tile = [&](int k0, int buf) {
#pragma unroll
        for (int i = 0; i < 2; ++i) {
            int slot = tid + 256 * i;          // 0..511 b128 transfers for A
            int r = slot >> 2;                 // tile row 0..127
            int c4 = slot & 3;                 // 16B chunk within 64B row
            int rg = rowBase + r;
            if (rg > M - 1) rg = M - 1;        // clamp: no divergence, rows>=M
                                               // produce garbage only in rows
                                               // that are never stored
            unsigned lds = (unsigned)(size_t)&AsAll[buf * 5120 + r * 40 + c4 * 8];
            unsigned long long ga =
                (unsigned long long)(A + (size_t)rg * K + k0 + c4 * 8);
            ASYNC_B128(lds, ga);
        }
        {
            int r = tid >> 2;                  // 0..63 B-tile rows (n)
            int c4 = tid & 3;
            unsigned lds = (unsigned)(size_t)&BsAll[buf * 2560 + r * 40 + c4 * 8];
            unsigned long long ga =
                (unsigned long long)(Bt + (size_t)(colBase + r) * K + k0 + c4 * 8);
            ASYNC_B128(lds, ga);
        }
    };

    const int nk = K >> 5;
    issue_tile(0, 0);

    for (int kt = 0; kt < nk; ++kt) {
        const int buf = kt & 1;
        // All waves have consumed buf^1 fragment reads from iteration kt-1
        // (each wave waits DScnt before its WMMAs, which precede this barrier).
        __syncthreads();
        if (kt + 1 < nk) {
            issue_tile((kt + 1) << 5, buf ^ 1);
            // per-wave FIFO: <=3 outstanding  =>  tile kt fully landed
            asm volatile("s_wait_asynccnt 0x3" ::: "memory");
        } else {
            asm volatile("s_wait_asynccnt 0x0" ::: "memory");
        }
        __syncthreads();   // tile kt visible to all waves

        const _Float16* Asb = &AsAll[buf * 5120];
        const _Float16* Bsb = &BsAll[buf * 2560];

        // ---- fragments per ISA 7.12.2 layouts ---------------------------
        union FragU { uint4 q[2]; v16h v; };
        FragU a0, a1, b0, b1;
        // A lane (m = l15, half hh): K runs [8h..8h+7], [16+8h..16+8h+7]
        const int ar0 = (wm * 32 + l15) * 40;
        a0.q[0] = *(const uint4*)&Asb[ar0 + hh * 8];
        a0.q[1] = *(const uint4*)&Asb[ar0 + 16 + hh * 8];
        const int ar1 = ar0 + 16 * 40;
        a1.q[0] = *(const uint4*)&Asb[ar1 + hh * 8];
        a1.q[1] = *(const uint4*)&Asb[ar1 + 16 + hh * 8];
        // B lane (n = l15, half hh): K runs [8h..8h+7], [16+8h..16+8h+7]
        const int br0 = (wn * 32 + l15) * 40;
        b0.q[0] = *(const uint4*)&Bsb[br0 + hh * 8];
        b0.q[1] = *(const uint4*)&Bsb[br0 + 16 + hh * 8];
        const int br1 = br0 + 16 * 40;
        b1.q[0] = *(const uint4*)&Bsb[br1 + hh * 8];
        b1.q[1] = *(const uint4*)&Bsb[br1 + 16 + hh * 8];

        c00 = __builtin_amdgcn_wmma_f32_16x16x32_f16(false, a0.v, false, b0.v,
                                                     (short)0, c00, false, false);
        c01 = __builtin_amdgcn_wmma_f32_16x16x32_f16(false, a0.v, false, b1.v,
                                                     (short)0, c01, false, false);
        c10 = __builtin_amdgcn_wmma_f32_16x16x32_f16(false, a1.v, false, b0.v,
                                                     (short)0, c10, false, false);
        c11 = __builtin_amdgcn_wmma_f32_16x16x32_f16(false, a1.v, false, b1.v,
                                                     (short)0, c11, false, false);
    }

    // ---- store: C/D layout  n = l15, m = vgpr + 8*(lane>=16) -------------
    const int cg0 = colBase + wn * 32 + l15;
    const int cg1 = cg0 + 16;
    const float bias0 = bias[cg0];
    const float bias1 = bias[cg1];
    const int rg = rowBase + wm * 32 + 8 * (lane >> 4);
#pragma unroll
    for (int i = 0; i < 8; ++i) {
        int r0 = rg + i;
        if (r0 < M) {
            Cout[(size_t)r0 * NCOLS + cg0] = c00[i] + bias0;
            Cout[(size_t)r0 * NCOLS + cg1] = c01[i] + bias1;
        }
        int r1 = r0 + 16;
        if (r1 < M) {
            Cout[(size_t)r1 * NCOLS + cg0] = c10[i] + bias0;
            Cout[(size_t)r1 * NCOLS + cg1] = c11[i] + bias1;
        }
    }
}

// ---------------------------------------------------------------------------
// Orchestration
// ---------------------------------------------------------------------------
extern "C" void kernel_launch(void* const* d_in, const int* in_sizes, int n_in,
                              void* d_out, int out_size, void* d_ws, size_t ws_size,
                              hipStream_t stream) {
    const float* x    = (const float*)d_in[0];
    const int*   ei   = (const int*)d_in[1];
    const float* bn_g = (const float*)d_in[2];
    const float* bn_b = (const float*)d_in[3];
    const float* W1   = (const float*)d_in[4];
    const float* b1   = (const float*)d_in[5];
    const float* g1   = (const float*)d_in[6];
    const float* be1  = (const float*)d_in[7];
    const float* W2   = (const float*)d_in[8];
    const float* b2   = (const float*)d_in[9];
    const float* g2   = (const float*)d_in[10];
    const float* be2  = (const float*)d_in[11];
    const float* W3   = (const float*)d_in[12];
    const float* b3   = (const float*)d_in[13];
    const float* We   = (const float*)d_in[14];
    const float* bev  = (const float*)d_in[15];
    float* out = (float*)d_out;

    const int C  = 128;
    const int Hh = 256;
    const int Nn = in_sizes[0] / C;
    const int Ee = in_sizes[1] / 2;

    float*    ws  = (float*)d_ws;
    _Float16* hws = (_Float16*)d_ws;
    const size_t NC = (size_t)Nn * C;

    // Aliased layout (~167 MB, mostly L2-resident on MI455X's 192 MB L2):
    //  floats [0,NC):     h (f32)            -> later h1_16 (2NC halfs)
    //  floats [NC,3NC):   mx, den            -> later t2 (f32)
    //  floats [3NC,4NC):  num                -> later h2_16 (2NC halfs)
    //  floats [4NC,6NC):  t1 (f32)           -> later t3 f32 in [4NC,5NC)
    //  floats [6NC,6.5NC): A16 (NC halfs)    -> later t3_16
    //  then f16 transposed weights (147456 halfs), then stats (1024 floats)
    float*    f_h   = ws;
    float*    f_mx  = ws + NC;
    float*    f_den = ws + 2 * NC;
    float*    f_num = ws + 3 * NC;
    float*    f_t1  = ws + 4 * NC;
    float*    f_t2  = f_mx;
    float*    f_t3  = f_t1;
    _Float16* h1_16 = hws;                  // halfs [0, 2NC)
    _Float16* h2_16 = hws + 6 * NC;         // halfs [6NC, 8NC)
    _Float16* A16   = hws + 12 * NC;        // halfs [12NC, 13NC)
    _Float16* t3_16 = A16;
    _Float16* W1t16 = hws + 13 * NC;
    _Float16* W2t16 = W1t16 + 32768;
    _Float16* W3t16 = W2t16 + 65536;
    _Float16* Wet16 = W3t16 + 32768;
    float* f_stats = ws + 6 * NC + NC / 2 + 73728;
    float* sums  = f_stats;
    float* sqs   = f_stats + 256;
    float* scale = f_stats + 512;
    float* shift = f_stats + 768;

    const dim3 gH(Hh / 64, (Nn + 127) / 128);
    const dim3 gC(C / 64, (Nn + 127) / 128);

    // --- zero atomic accumulators; convert+transpose weights to f16 -------
    zero_kernel<<<1024, 256, 0, stream>>>(f_mx, 3 * NC);
    transpose_f16_kernel<<<(C * Hh + 255) / 256, 256, 0, stream>>>(W1, W1t16, C, Hh);
    transpose_f16_kernel<<<(Hh * Hh + 255) / 256, 256, 0, stream>>>(W2, W2t16, Hh, Hh);
    transpose_f16_kernel<<<(Hh * C + 255) / 256, 256, 0, stream>>>(W3, W3t16, Hh, C);
    transpose_f16_kernel<<<(C * C + 255) / 256, 256, 0, stream>>>(We, Wet16, C, C);

    // --- outer BatchNorm + ReLU over x (f32 h feeds edge kernels) ---------
    zero_kernel<<<4, 256, 0, stream>>>(sums, 512);
    colstats_kernel<<<512, C, 0, stream>>>(x, Nn, C, sums, sqs);
    finalize_stats_kernel<<<1, C, 0, stream>>>(sums, sqs, bn_g, bn_b, Nn, scale, shift);
    bn_relu_kernel<<<2048, 256, 0, stream>>>(x, f_h, scale, shift, NC, C - 1);

    // --- edge softmax aggregation -----------------------------------------
    edge_max_kernel<<<2048, C, 0, stream>>>(f_h, ei, Ee, (int*)f_mx);
    edge_sum_kernel<<<2048, C, 0, stream>>>(f_h, ei, Ee, (const int*)f_mx, f_den, f_num);
    node_combine_f16_kernel<<<2048, 256, 0, stream>>>(f_h, f_num, f_den, A16, NC);

    // --- MLP: Linear(C,H) -> BN -> ReLU -----------------------------------
    gemm_wmma_kernel<<<gH, 256, 0, stream>>>(A16, W1t16, b1, f_t1, Nn, C, Hh);
    zero_kernel<<<4, 256, 0, stream>>>(sums, 512);
    colstats_kernel<<<512, Hh, 0, stream>>>(f_t1, Nn, Hh, sums, sqs);
    finalize_stats_kernel<<<1, Hh, 0, stream>>>(sums, sqs, g1, be1, Nn, scale, shift);
    bn_relu_f16_kernel<<<2048, 256, 0, stream>>>(f_t1, h1_16, scale, shift, 2 * NC, Hh - 1);

    // --- Linear(H,H) -> BN -> ReLU ----------------------------------------
    gemm_wmma_kernel<<<gH, 256, 0, stream>>>(h1_16, W2t16, b2, f_t2, Nn, Hh, Hh);
    zero_kernel<<<4, 256, 0, stream>>>(sums, 512);
    colstats_kernel<<<512, Hh, 0, stream>>>(f_t2, Nn, Hh, sums, sqs);
    finalize_stats_kernel<<<1, Hh, 0, stream>>>(sums, sqs, g2, be2, Nn, scale, shift);
    bn_relu_f16_kernel<<<2048, 256, 0, stream>>>(f_t2, h2_16, scale, shift, 2 * NC, Hh - 1);

    // --- Linear(H,C) + residual -------------------------------------------
    gemm_wmma_kernel<<<gC, 256, 0, stream>>>(h2_16, W3t16, b3, f_t3, Nn, Hh, C);
    add_f16_kernel<<<2048, 256, 0, stream>>>(f_t3, x, t3_16, NC);

    // --- encoder Linear(C,OUT) -> d_out -----------------------------------
    gemm_wmma_kernel<<<gC, 256, 0, stream>>>(t3_16, Wet16, bev, out, Nn, C, C);
}
